// RelativeMultiHeadAttention_5660766896264
// MI455X (gfx1250) — matrix operand order repaired and checked by reading, not compiled, not run on hardware
//
#include <hip/hip_runtime.h>
#include <stdint.h>
#include <math.h>

// ---------------------------------------------------------------------------
// RelativeMultiHeadAttention for MI455X (gfx1250, wave32, WMMA bf16)
//   out = softmax( (XWq)(XWk)^T * scale + relbias ) (XWv)  @ Wo^T
// 5 stages: weight fp32->bf16 convert, 3x projection GEMM (wmma),
//           flash-attention (wmma), output GEMM (wmma).
// ---------------------------------------------------------------------------

typedef __attribute__((ext_vector_type(16))) __bf16 v16bf;
typedef __attribute__((ext_vector_type(8)))  float  v8f;

#define S_LEN   2048
#define BATCH   2
#define HEADS   16
#define DK      64
#define DMODEL  1024
#define MAXLEN  2048
#define SCALE_F 0.125f  // 1/sqrt(64), exact power of two

union V16U {
  v16bf  v;
  __bf16 h[16];
  uint4  q[2];
};

__device__ __forceinline__ __bf16 f2bf(float f) {
  union { float f; uint32_t u; } a; a.f = f;
  uint32_t r = a.u + 0x7FFFu + ((a.u >> 16) & 1u);   // round-to-nearest-even
  union { uint16_t s; __bf16 b; } o; o.s = (uint16_t)(r >> 16);
  return o.b;
}

// ---------------------------------------------------------------------------
// fp32 -> bf16 streaming conversion (for the 4 weight matrices)
// ---------------------------------------------------------------------------
__global__ void __launch_bounds__(256)
cvt_f32_bf16(const float* __restrict__ in, __bf16* __restrict__ out, int n)
{
  const int i = (blockIdx.x * 256 + threadIdx.x) * 8;
  if (i + 8 <= n) {
    const float4 a = *(const float4*)(in + i);
    const float4 b = *(const float4*)(in + i + 4);
    union { __bf16 h[8]; uint4 q; } o;
    o.h[0] = f2bf(a.x); o.h[1] = f2bf(a.y); o.h[2] = f2bf(a.z); o.h[3] = f2bf(a.w);
    o.h[4] = f2bf(b.x); o.h[5] = f2bf(b.y); o.h[6] = f2bf(b.z); o.h[7] = f2bf(b.w);
    *(uint4*)(out + i) = o.q;
  }
}

// ---------------------------------------------------------------------------
// GEMM: C[M,N] = A[M,K] * W[N,K]^T + bias[N]   (W already bf16)
//   A_BF16=false : A fp32 (converted in-register) ; true : A bf16
//   MODE=0 : write bf16 into [b,h,s,d] layout, scaled by oscale (projections)
//   MODE=1 : write fp32 row-major M x N (final output)
// block = 128 (4 waves); wave computes 32x64 C tile; block = 128x64.
// ---------------------------------------------------------------------------
template <bool A_BF16, int MODE>
__global__ void __launch_bounds__(128)
gemm_wmma(const void* __restrict__ Ain,
          const __bf16* __restrict__ W,
          const float* __restrict__ bias,
          void* __restrict__ Out,
          int M, int N, int K, float oscale)
{
  const int lane  = threadIdx.x & 31;
  const int wave  = threadIdx.x >> 5;
  const int lh    = lane >> 4;        // lane half (0/1)
  const int ll    = lane & 15;
  const int mbase = (blockIdx.x * 4 + wave) * 32;
  const int nbase = blockIdx.y * 64;

  v8f acc[2][4] = {};

  for (int kb = 0; kb < K; kb += 32) {
    // ---- A fragments (2 M-subtiles): row = ll, K chunks at lh*8 / lh*8+16
    V16U a[2];
    const int k0 = kb + lh * 8;
#pragma unroll
    for (int mt = 0; mt < 2; ++mt) {
      const int am = mbase + mt * 16 + ll;
      if constexpr (A_BF16) {
        const __bf16* ap = (const __bf16*)Ain + (size_t)am * K;
        a[mt].q[0] = *(const uint4*)(ap + k0);
        a[mt].q[1] = *(const uint4*)(ap + k0 + 16);
      } else {
        const float* ap = (const float*)Ain + (size_t)am * K;
#pragma unroll
        for (int j = 0; j < 8; ++j) a[mt].h[j]     = f2bf(ap[k0 + j]);
#pragma unroll
        for (int j = 0; j < 8; ++j) a[mt].h[8 + j] = f2bf(ap[k0 + 16 + j]);
      }
    }
    // ---- B fragments: lane holds W row n, K chunks at lh*8 / lh*8+16
    const int k0b = kb + lh * 8;
#pragma unroll
    for (int nt = 0; nt < 4; ++nt) {
      const __bf16* wp = W + (size_t)(nbase + nt * 16 + ll) * K + k0b;
      V16U b;
      b.q[0] = *(const uint4*)(wp);
      b.q[1] = *(const uint4*)(wp + 16);
      if (kb + 32 < K) __builtin_prefetch(wp + 32, 0, 1);  // next K chunk
      acc[0][nt] = __builtin_amdgcn_wmma_f32_16x16x32_bf16(
          false, a[0].v, false, b.v, (short)0, acc[0][nt], false, false);
      acc[1][nt] = __builtin_amdgcn_wmma_f32_16x16x32_bf16(
          false, a[1].v, false, b.v, (short)0, acc[1][nt], false, false);
    }
  }

  // ---- epilogue: C layout row = r + 8*lh, col = ll (per 16-wide tile)
#pragma unroll
  for (int nt = 0; nt < 4; ++nt) {
    const int   n  = nbase + nt * 16 + ll;
    const float bv = bias[n];
#pragma unroll
    for (int mt = 0; mt < 2; ++mt) {
#pragma unroll
      for (int r = 0; r < 8; ++r) {
        const int   m   = mbase + mt * 16 + r + 8 * lh;
        const float val = acc[mt][nt][r] + bv;
        if constexpr (MODE == 0) {
          // m = s*BATCH + b ; n = h*DK + d  ->  [b][h][s][d] bf16 (scaled)
          const int s  = m >> 1, bb = m & 1;
          const int hh = n >> 6, dd = n & 63;
          ((__bf16*)Out)[(((size_t)(bb * HEADS + hh) * S_LEN + s) << 6) + dd] =
              f2bf(val * oscale);
        } else {
          ((float*)Out)[(size_t)m * N + n] = val;
        }
      }
    }
  }
}

// ---------------------------------------------------------------------------
// Flash attention per (b,h): block = 128 (4 waves), 64 q rows / block,
// wave owns 16 q rows. Loop over 64-key tiles with online softmax.
// Q is pre-scaled by 1/sqrt(dk). rel_table column for this head is staged
// into LDS once per block (distance window = [qb0, qb0+2110]).
// ---------------------------------------------------------------------------
__global__ void __launch_bounds__(128)
attn_fa_wmma(const __bf16* __restrict__ Qm,
             const __bf16* __restrict__ Km,
             const __bf16* __restrict__ Vm,
             const float*  __restrict__ rel,
             __bf16* __restrict__ outf)
{
  __shared__ __align__(16) __bf16 VT[DK][72];       // V^T tile: [d][key]
  __shared__ __align__(16) __bf16 PB[4][16][72];    // per-wave P (16q x 64k)
  __shared__ __align__(16) float  RELS[2112];       // rel bias window, 1 head

  const int lane  = threadIdx.x & 31;
  const int wave  = threadIdx.x >> 5;
  const int lh    = lane >> 4;
  const int ll    = lane & 15;
  const int bh    = blockIdx.y;                 // b*HEADS + h
  const int hh    = bh & (HEADS - 1);
  const int bb    = bh >> 4;
  const int qb0   = blockIdx.x * 64;
  const int qbase = qb0 + wave * 16;

  const __bf16* Qh = Qm + (size_t)bh * S_LEN * DK;
  const __bf16* Kh = Km + (size_t)bh * S_LEN * DK;
  const __bf16* Vh = Vm + (size_t)bh * S_LEN * DK;

  // stage rel-bias window: dist = q - k + 2047 = qb0 + idx, idx in [0,2110]
  for (int i = threadIdx.x; i < 2111; i += 128)
    RELS[i] = rel[(size_t)(qb0 + i) * HEADS + hh];

  // Q fragments (A-matrix layout), resident for whole kernel
  V16U aq[2];
  {
    const __bf16* qp = Qh + (size_t)(qbase + ll) * DK;
#pragma unroll
    for (int kk = 0; kk < 2; ++kk) {
      const int d0 = kk * 32 + lh * 8;
      aq[kk].q[0] = *(const uint4*)(qp + d0);
      aq[kk].q[1] = *(const uint4*)(qp + d0 + 16);
    }
  }

  v8f   acc[4] = {};
  float m_i[8], l_i[8];
#pragma unroll
  for (int r = 0; r < 8; ++r) { m_i[r] = -INFINITY; l_i[r] = 0.f; }

  for (int kt = 0; kt < S_LEN / 64; ++kt) {
    const int kbase = kt * 64;
    __syncthreads();                       // VT reuse + (iter 0) RELS staging
    {                                      // stage V^T cooperatively
      const int key = threadIdx.x & 63;
      const int dh  = threadIdx.x >> 6;    // 0/1 -> d half
      const __bf16* vp = Vh + (size_t)(kbase + key) * DK + dh * 32;
#pragma unroll
      for (int j = 0; j < 32; ++j) VT[dh * 32 + j][key] = vp[j];
    }
    __syncthreads();

    // ---- GEMM1: S = Q K^T (Kdim = d = 64)
    v8f sc[4];
#pragma unroll
    for (int nt = 0; nt < 4; ++nt) {
      const __bf16* kp = Kh + (size_t)(kbase + nt * 16 + ll) * DK;
      const int d0 = lh * 8;               // B layout: K chunks at lh*8 / lh*8+16
      V16U bk0, bk1;
      bk0.q[0] = *(const uint4*)(kp + d0);
      bk0.q[1] = *(const uint4*)(kp + d0 + 16);
      bk1.q[0] = *(const uint4*)(kp + 32 + d0);
      bk1.q[1] = *(const uint4*)(kp + 32 + d0 + 16);
      v8f s = {};
      s = __builtin_amdgcn_wmma_f32_16x16x32_bf16(false, aq[0].v, false, bk0.v,
                                                  (short)0, s, false, false);
      s = __builtin_amdgcn_wmma_f32_16x16x32_bf16(false, aq[1].v, false, bk1.v,
                                                  (short)0, s, false, false);
      sc[nt] = s;
    }

    // ---- relative position bias (Q already carries 1/sqrt(dk))
#pragma unroll
    for (int nt = 0; nt < 4; ++nt) {
      const int idxbase = wave * 16 + 8 * lh - kbase - nt * 16 - ll + (MAXLEN - 1);
#pragma unroll
      for (int r = 0; r < 8; ++r)
        sc[nt][r] += RELS[idxbase + r];
    }

    // ---- online softmax: row reductions across the 16-lane half
    float mt[8];
#pragma unroll
    for (int r = 0; r < 8; ++r)
      mt[r] = fmaxf(fmaxf(sc[0][r], sc[1][r]), fmaxf(sc[2][r], sc[3][r]));
#pragma unroll
    for (int mask = 1; mask < 16; mask <<= 1)
#pragma unroll
      for (int r = 0; r < 8; ++r)
        mt[r] = fmaxf(mt[r], __shfl_xor(mt[r], mask, 32));

    float alpha[8], rs[8];
#pragma unroll
    for (int r = 0; r < 8; ++r) {
      const float mn = fmaxf(m_i[r], mt[r]);
      alpha[r] = __expf(m_i[r] - mn);
      m_i[r]   = mn;
      rs[r]    = 0.f;
    }
#pragma unroll
    for (int nt = 0; nt < 4; ++nt)
#pragma unroll
      for (int r = 0; r < 8; ++r) {
        const float p = __expf(sc[nt][r] - m_i[r]);
        rs[r] += p;
        PB[wave][r + 8 * lh][nt * 16 + ll] = f2bf(p);   // C-layout -> LDS
      }
#pragma unroll
    for (int mask = 1; mask < 16; mask <<= 1)
#pragma unroll
      for (int r = 0; r < 8; ++r)
        rs[r] += __shfl_xor(rs[r], mask, 32);
#pragma unroll
    for (int r = 0; r < 8; ++r) l_i[r] = l_i[r] * alpha[r] + rs[r];
#pragma unroll
    for (int nt = 0; nt < 4; ++nt)
#pragma unroll
      for (int r = 0; r < 8; ++r) acc[nt][r] *= alpha[r];

    // ---- GEMM2: acc += P @ V  (Kdim = 64 keys)
#pragma unroll
    for (int kk = 0; kk < 2; ++kk) {
      V16U ap;                                    // A-layout read of P from LDS
      const int kb2 = kk * 32 + lh * 8;
      ap.q[0] = *(const uint4*)(&PB[wave][ll][kb2]);
      ap.q[1] = *(const uint4*)(&PB[wave][ll][kb2 + 16]);
      const int ks = kk * 32 + lh * 8;
#pragma unroll
      for (int ndt = 0; ndt < 4; ++ndt) {
        V16U bv;                                  // B-layout read of V^T
        bv.q[0] = *(const uint4*)(&VT[ndt * 16 + ll][ks]);
        bv.q[1] = *(const uint4*)(&VT[ndt * 16 + ll][ks + 16]);
        acc[ndt] = __builtin_amdgcn_wmma_f32_16x16x32_bf16(
            false, ap.v, false, bv.v, (short)0, acc[ndt], false, false);
      }
    }
  }

  // ---- epilogue: /l, write bf16 at row = s*B+b, col = h*64+d
#pragma unroll
  for (int ndt = 0; ndt < 4; ++ndt)
#pragma unroll
    for (int r = 0; r < 8; ++r) {
      const int   q   = qbase + r + 8 * lh;
      const int   d   = ndt * 16 + ll;
      const float val = acc[ndt][r] / l_i[r];
      outf[(size_t)(q * BATCH + bb) * DMODEL + hh * DK + d] = f2bf(val);
    }
}

// ---------------------------------------------------------------------------
extern "C" void kernel_launch(void* const* d_in, const int* in_sizes, int n_in,
                              void* d_out, int out_size, void* d_ws, size_t ws_size,
                              hipStream_t stream)
{
  const float* query = (const float*)d_in[0];
  const float* key   = (const float*)d_in[1];
  const float* value = (const float*)d_in[2];
  const float* Wq = (const float*)d_in[3];  const float* bq = (const float*)d_in[4];
  const float* Wk = (const float*)d_in[5];  const float* bk = (const float*)d_in[6];
  const float* Wv = (const float*)d_in[7];  const float* bv = (const float*)d_in[8];
  const float* Wo = (const float*)d_in[9];  const float* bo = (const float*)d_in[10];
  const float* rel = (const float*)d_in[11];

  const int M = S_LEN * BATCH;   // 4096
  const int N = DMODEL;          // 1024
  const int K = DMODEL;          // 1024
  const int WN = DMODEL * DMODEL;

  char* ws = (char*)d_ws;                                 // 40 MB used
  __bf16* wqb = (__bf16*)(ws);                            // 2 MB bf16 weights
  __bf16* wkb = (__bf16*)(ws + ((size_t)2 << 20));
  __bf16* wvb = (__bf16*)(ws + ((size_t)4 << 20));
  __bf16* wob = (__bf16*)(ws + ((size_t)6 << 20));
  __bf16* qbuf = (__bf16*)(ws + ((size_t)8  << 20));      // 8 MB [b,h,s,d]
  __bf16* kbuf = (__bf16*)(ws + ((size_t)16 << 20));      // 8 MB
  __bf16* vbuf = (__bf16*)(ws + ((size_t)24 << 20));      // 8 MB
  __bf16* abuf = (__bf16*)(ws + ((size_t)32 << 20));      // 8 MB (s*B+b, h*64+d)

  // stage 0: weight conversion (streaming, ~16 MB read)
  dim3 cblk(256), cgrid(WN / (256 * 8));
  cvt_f32_bf16<<<cgrid, cblk, 0, stream>>>(Wq, wqb, WN);
  cvt_f32_bf16<<<cgrid, cblk, 0, stream>>>(Wk, wkb, WN);
  cvt_f32_bf16<<<cgrid, cblk, 0, stream>>>(Wv, wvb, WN);
  cvt_f32_bf16<<<cgrid, cblk, 0, stream>>>(Wo, wob, WN);

  // stages 1-3: projections (Q pre-scaled by 1/sqrt(dk))
  dim3 gblk(128), ggrid(M / 128, N / 64);
  gemm_wmma<false, 0><<<ggrid, gblk, 0, stream>>>(query, wqb, bq, qbuf, M, N, K, SCALE_F);
  gemm_wmma<false, 0><<<ggrid, gblk, 0, stream>>>(key,   wkb, bk, kbuf, M, N, K, 1.0f);
  gemm_wmma<false, 0><<<ggrid, gblk, 0, stream>>>(value, wvb, bv, vbuf, M, N, K, 1.0f);

  // stage 4: flash attention with relative bias
  dim3 agrid(S_LEN / 64, BATCH * HEADS);
  attn_fa_wmma<<<agrid, gblk, 0, stream>>>(qbuf, kbuf, vbuf, rel, abuf);

  // stage 5: output projection, fp32 result
  gemm_wmma<true, 1><<<ggrid, gblk, 0, stream>>>(abuf, wob, bo, d_out, M, N, K, 1.0f);
}
